// MultiHeadSelfAttention_27470610825618
// MI455X (gfx1250) — matrix operand order repaired and checked by reading, not compiled, not run on hardware
//
#include <hip/hip_runtime.h>
#include <hip/hip_bf16.h>

#define NHEADS   16
#define HDIM     64
#define DMODEL   1024
#define SEQ      2048
#define BATCH    2
#define ATTN_SCALE 0.125f   // 1/sqrt(64)
#define KCHUNKS  (DMODEL / 32)

#define A_BUF_DW (256 * 20)
#define B_BUF_DW (64 * 20)

typedef unsigned int   u32b;
typedef unsigned short u16b;
typedef __attribute__((ext_vector_type(16))) __bf16 v16bf;
typedef __attribute__((ext_vector_type(8)))  float  v8f;
typedef __attribute__((ext_vector_type(8)))  u32b   v8u;

__device__ inline u16b f2bf(float f) {  // RNE fp32 -> bf16 bits
    u32b u = __float_as_uint(f);
    return (u16b)((u + 0x7FFFu + ((u >> 16) & 1u)) >> 16);
}

// A-fragment (16x32 bf16) K-start for VGPR r, lane-half g (contiguous: k = 16g + 2r)
__device__ inline int a_k0(int r, int g) {
    return g * 16 + 2 * r;
}

__device__ inline v8f wmma_bf16(v16bf a, v16bf b, v8f c) {
    return __builtin_amdgcn_wmma_f32_16x16x32_bf16(false, a, false, b, (short)0, c,
                                                   false, false);
}

// Async global->LDS 16B copy per lane (CDNA5, tracked by ASYNCcnt).
// LDS dest = low 32 bits of the flat shared-space address (ISA 10.2: LDS
// apertures map via addr[31:0]).
__device__ inline void async_ld_b128(void* lds, const void* g) {
    asm volatile("global_load_async_to_lds_b128 %0, %1, off"
                 :: "v"((u32b)(uintptr_t)lds),
                    "v"((unsigned long long)(uintptr_t)g)
                 : "memory");
}
__device__ inline void wait_async0() {
    asm volatile("s_wait_asynccnt 0x0" ::: "memory");
}

// ---------------------------------------------------------------- convert
__global__ __launch_bounds__(256) void cvt_f32_bf16(const float* __restrict__ src,
                                                    u16b* __restrict__ dst, int n4) {
    int i = blockIdx.x * blockDim.x + threadIdx.x;
    if (i < n4) {
        float4 f = ((const float4*)src)[i];
        uint2 o;
        o.x = (u32b)f2bf(f.x) | ((u32b)f2bf(f.y) << 16);
        o.y = (u32b)f2bf(f.z) | ((u32b)f2bf(f.w) << 16);
        ((uint2*)dst)[i] = o;
    }
}

// -------------------------------------------------- shared GEMM mainloop
// Block tile 256(M) x 64(N), K-chunks of 32. 8 waves, each wave owns 32 rows
// (2 A-frags x 4 B-frags = 8 WMMAs/chunk). Tiles staged by the async copy
// engine into double-buffered LDS: zero staging VGPRs, one barrier/chunk.
__device__ inline void async_tile_stage(const u32b* __restrict__ Ag,
                                        const u32b* __restrict__ Wg,
                                        u32b* dstA, u32b* dstB,
                                        int m0, int n0, int kk,
                                        int tid, int brow, int bseg) {
#pragma unroll
    for (int j = 0; j < 4; ++j)
        async_ld_b128(dstA + tid * 20 + j * 4,
                      Ag + (size_t)(m0 + tid) * 512 + kk * 16 + j * 4);
    async_ld_b128(dstB + brow * 20 + bseg * 4,
                  Wg + (size_t)(n0 + brow) * 512 + kk * 16 + bseg * 4);
}

__device__ inline void gemm_mainloop(const u32b* __restrict__ Ag,
                                     const u32b* __restrict__ Wg,
                                     u32b* ldsA, u32b* ldsB,   // double buffers
                                     int m0, int n0, v8f acc[2][4],
                                     int tid, int wv, int n, int g) {
    const int brow = tid >> 2, bseg = tid & 3;

    async_tile_stage(Ag, Wg, ldsA, ldsB, m0, n0, 0, tid, brow, bseg);
    wait_async0();
    __syncthreads();   // chunk 0 visible block-wide

    for (int kk = 0; kk < KCHUNKS; ++kk) {
        u32b* curA = ldsA + (kk & 1) * A_BUF_DW;
        u32b* curB = ldsB + (kk & 1) * B_BUF_DW;
        if (kk + 1 < KCHUNKS)   // async engine fills other buffer during WMMAs
            async_tile_stage(Ag, Wg, ldsA + ((kk + 1) & 1) * A_BUF_DW,
                             ldsB + ((kk + 1) & 1) * B_BUF_DW,
                             m0, n0, kk + 1, tid, brow, bseg);

        v16bf af[2];
#pragma unroll
        for (int s2 = 0; s2 < 2; ++s2) {
            v8u au;
#pragma unroll
            for (int r = 0; r < 8; ++r)
                au[r] = curA[(wv * 32 + s2 * 16 + n) * 20 + (a_k0(r, g) >> 1)];
            af[s2] = __builtin_bit_cast(v16bf, au);
        }
#pragma unroll
        for (int t = 0; t < 4; ++t) {
            v8u bu;
#pragma unroll
            for (int r = 0; r < 8; ++r) bu[r] = curB[(t * 16 + n) * 20 + g * 8 + r];
            v16bf bfr = __builtin_bit_cast(v16bf, bu);
#pragma unroll
            for (int s2 = 0; s2 < 2; ++s2)
                acc[s2][t] = wmma_bf16(af[s2], bfr, acc[s2][t]);
        }

        if (kk + 1 < KCHUNKS) wait_async0();  // own next-chunk writes landed
        __syncthreads();   // all waves: cur reads done + next chunk visible
    }
}

// ------------------------------------------------- GEMM1: QKV + bias + RoPE
__global__ __launch_bounds__(256) void gemm_qkv_rope(
    const u16b* __restrict__ Xbf, const u16b* __restrict__ Wbf,
    const float* __restrict__ bias,
    u16b* __restrict__ Qb, u16b* __restrict__ Kb, u16b* __restrict__ Vb) {
    __shared__ u32b ldsA[2 * A_BUF_DW];
    __shared__ u32b ldsB[2 * B_BUF_DW];

    const int tid = threadIdx.x, lane = tid & 31, wv = tid >> 5;
    const int n = lane & 15, g = lane >> 4;
    const int n0 = blockIdx.x * 64;
    const int m0 = blockIdx.y * 256;

    v8f acc[2][4];
#pragma unroll
    for (int s2 = 0; s2 < 2; ++s2)
#pragma unroll
        for (int t = 0; t < 4; ++t)
            acc[s2][t] = (v8f){0.f, 0.f, 0.f, 0.f, 0.f, 0.f, 0.f, 0.f};

    gemm_mainloop((const u32b*)Xbf, (const u32b*)Wbf, ldsA, ldsB, m0, n0, acc,
                  tid, wv, n, g);

    // epilogue: bias + RoPE + scatter to [B,H,S,64] (part/head uniform per block)
    const int part = n0 / DMODEL;               // 0=q 1=k 2=v
    const int h = (n0 % DMODEL) / HDIM;
#pragma unroll
    for (int t = 0; t < 4; ++t) {
        const int d = t * 16 + n;               // head-dim coord 0..63
        const float bv = bias[n0 + t * 16 + n];
        const int j = d >> 1;
        const float inv = __expf(-(float)(2 * j) * 0.14391157f);  // ln(10000)/64
#pragma unroll
        for (int s2 = 0; s2 < 2; ++s2) {
#pragma unroll
            for (int r = 0; r < 8; ++r) {
                int row = m0 + wv * 32 + s2 * 16 + g * 8 + r;
                int bb = row >> 11, s = row & 2047;
                float val = acc[s2][t][r] + bv;
                size_t idx = ((size_t)(bb * NHEADS + h) * SEQ + s) * HDIM + d;
                if (part == 2) {
                    Vb[idx] = f2bf(val);
                } else {
                    float partner = __shfl_xor(val, 1, 32);  // pair element (d^1)
                    float ang = (float)s * inv;
                    float cs = __cosf(ang), sn = __sinf(ang);
                    float o = (d & 1) ? (partner * sn + val * cs)
                                      : (val * cs - partner * sn);
                    if (part == 0) Qb[idx] = f2bf(o * ATTN_SCALE);
                    else           Kb[idx] = f2bf(o);
                }
            }
        }
    }
}

// -------------------------------------------------------- flash attention
__global__ __launch_bounds__(256) void attention(
    const u16b* __restrict__ Qb, const u16b* __restrict__ Kb,
    const u16b* __restrict__ Vb, const unsigned char* __restrict__ mask,
    u16b* __restrict__ attn) {
    __shared__ u32b ldsK[32 * 36];       // [key][d] 32x64 bf16, stride 36 dwords
    __shared__ u16b ldsVt[64 * 36];      // [d][key] transposed, stride 36 u16
    __shared__ u16b ldsP[8 * 16 * 40];   // per-wave P tiles 16x32, stride 40 u16

    const int tid = threadIdx.x, lane = tid & 31, wv = tid >> 5;
    const int n = lane & 15, g = lane >> 4;
    const int q0 = blockIdx.x * 128;
    const int h = blockIdx.y, bb = blockIdx.z;
    const int bh = bb * NHEADS + h;

    const u32b* Qg = (const u32b*)Qb + (size_t)bh * SEQ * 32;  // 32 dwords/row
    const u32b* Kg = (const u32b*)Kb + (size_t)bh * SEQ * 32;
    const u32b* Vg = (const u32b*)Vb + (size_t)bh * SEQ * 32;

    // resident Q fragments (2 K-chunks of 32 over head dim 64), scale pre-folded
    v16bf qf[2];
    {
        int qrow = q0 + wv * 16 + n;
#pragma unroll
        for (int c = 0; c < 2; ++c) {
            v8u u;
#pragma unroll
            for (int r = 0; r < 8; ++r)
                u[r] = Qg[(size_t)qrow * 32 + ((c * 32 + a_k0(r, g)) >> 1)];
            qf[c] = __builtin_bit_cast(v16bf, u);
        }
    }

    v8f acc[4];
#pragma unroll
    for (int t = 0; t < 4; ++t) acc[t] = (v8f){0.f, 0.f, 0.f, 0.f, 0.f, 0.f, 0.f, 0.f};
    float mrun[8], lrun[8];
#pragma unroll
    for (int r = 0; r < 8; ++r) { mrun[r] = -__builtin_inff(); lrun[r] = 0.f; }

    // register double-buffer for K/V chunk staging (V needs a transpose)
    const int krow = tid >> 3, kcol = (tid & 7) * 4;
    uint4 pk = *(const uint4*)(Kg + (size_t)krow * 32 + kcol);
    uint4 pv = *(const uint4*)(Vg + (size_t)krow * 32 + kcol);

    for (int key0 = 0; key0 < SEQ; key0 += 32) {
        __syncthreads();
        *(uint4*)(ldsK + krow * 36 + kcol) = pk;
        {
            u32b vvv[4] = {pv.x, pv.y, pv.z, pv.w};
#pragma unroll
            for (int jj = 0; jj < 4; ++jj) {
                int d = (kcol + jj) * 2;
                ldsVt[d * 36 + krow]       = (u16b)(vvv[jj] & 0xFFFFu);
                ldsVt[(d + 1) * 36 + krow] = (u16b)(vvv[jj] >> 16);
            }
        }
        __syncthreads();
        if (key0 + 32 < SEQ) {   // prefetch next chunk; overlaps compute below
            pk = *(const uint4*)(Kg + (size_t)(key0 + 32 + krow) * 32 + kcol);
            pv = *(const uint4*)(Vg + (size_t)(key0 + 32 + krow) * 32 + kcol);
        }

        // scores: S = Q @ K^T  (two 16x16 key tiles, K-dim 64 = 2 chunks)
        v8f sc[2];
        sc[0] = (v8f){0.f, 0.f, 0.f, 0.f, 0.f, 0.f, 0.f, 0.f};
        sc[1] = (v8f){0.f, 0.f, 0.f, 0.f, 0.f, 0.f, 0.f, 0.f};
#pragma unroll
        for (int c = 0; c < 2; ++c) {
#pragma unroll
            for (int t = 0; t < 2; ++t) {
                v8u bu;
#pragma unroll
                for (int r = 0; r < 8; ++r)
                    bu[r] = ldsK[(t * 16 + n) * 36 + c * 16 + g * 8 + r];
                v16bf bfr = __builtin_bit_cast(v16bf, bu);
                sc[t] = wmma_bf16(qf[c], bfr, sc[t]);
            }
        }

        unsigned char mv0 = mask[(size_t)bb * SEQ + key0 + n];
        unsigned char mv1 = mask[(size_t)bb * SEQ + key0 + 16 + n];

        // online softmax: rows live in 16-lane halves of the wave
#pragma unroll
        for (int r = 0; r < 8; ++r) {
            float s0 = mv0 ? sc[0][r] : -__builtin_inff();
            float s1 = mv1 ? sc[1][r] : -__builtin_inff();
            float mx = fmaxf(s0, s1);
#pragma unroll
            for (int off = 1; off < 16; off <<= 1) mx = fmaxf(mx, __shfl_xor(mx, off, 32));
            float mnew = fmaxf(mrun[r], mx);
            float alpha = __expf(mrun[r] - mnew);
            float p0 = __expf(s0 - mnew);
            float p1 = __expf(s1 - mnew);
            float ls = p0 + p1;
#pragma unroll
            for (int off = 1; off < 16; off <<= 1) ls += __shfl_xor(ls, off, 32);
            lrun[r] = lrun[r] * alpha + ls;
            mrun[r] = mnew;
#pragma unroll
            for (int t = 0; t < 4; ++t) acc[t][r] = acc[t][r] * alpha;
            int m = g * 8 + r;
            ldsP[(wv * 16 + m) * 40 + n]      = f2bf(p0);
            ldsP[(wv * 16 + m) * 40 + 16 + n] = f2bf(p1);
        }

        // wave-local LDS ordering: P writes above -> P reads below (same wave)
        asm volatile("s_wait_dscnt 0x0" ::: "memory");

        v8u pu;
#pragma unroll
        for (int r = 0; r < 8; ++r)
            pu[r] = *(const u32b*)&ldsP[(wv * 16 + n) * 40 + a_k0(r, g)];
        v16bf pf = __builtin_bit_cast(v16bf, pu);

#pragma unroll
        for (int t = 0; t < 4; ++t) {   // O += P @ V
            v8u bu;
#pragma unroll
            for (int r = 0; r < 8; ++r)
                bu[r] = *(const u32b*)&ldsVt[(t * 16 + n) * 36 + g * 16 + 2 * r];
            v16bf bfr = __builtin_bit_cast(v16bf, bu);
            acc[t] = wmma_bf16(pf, bfr, acc[t]);
        }
    }

    // normalize + scatter back to [B,S,H*64] bf16 for the output GEMM
#pragma unroll
    for (int t = 0; t < 4; ++t) {
#pragma unroll
        for (int r = 0; r < 8; ++r) {
            int qrow = q0 + wv * 16 + g * 8 + r;
            float o = acc[t][r] / lrun[r];
            attn[((size_t)bb * SEQ + qrow) * DMODEL + h * HDIM + t * 16 + n] = f2bf(o);
        }
    }
}

// ------------------------------------------------- GEMM2: output projection
__global__ __launch_bounds__(256) void gemm_out(
    const u16b* __restrict__ Abf, const u16b* __restrict__ Wbf,
    const float* __restrict__ bias, float* __restrict__ out) {
    __shared__ u32b ldsA[2 * A_BUF_DW];
    __shared__ u32b ldsB[2 * B_BUF_DW];

    const int tid = threadIdx.x, lane = tid & 31, wv = tid >> 5;
    const int n = lane & 15, g = lane >> 4;
    const int n0 = blockIdx.x * 64;
    const int m0 = blockIdx.y * 256;

    v8f acc[2][4];
#pragma unroll
    for (int s2 = 0; s2 < 2; ++s2)
#pragma unroll
        for (int t = 0; t < 4; ++t)
            acc[s2][t] = (v8f){0.f, 0.f, 0.f, 0.f, 0.f, 0.f, 0.f, 0.f};

    gemm_mainloop((const u32b*)Abf, (const u32b*)Wbf, ldsA, ldsB, m0, n0, acc,
                  tid, wv, n, g);

#pragma unroll
    for (int t = 0; t < 4; ++t) {
        float bv = bias[n0 + t * 16 + n];
#pragma unroll
        for (int s2 = 0; s2 < 2; ++s2) {
#pragma unroll
            for (int r = 0; r < 8; ++r) {
                int row = m0 + wv * 32 + s2 * 16 + g * 8 + r;
                out[(size_t)row * DMODEL + n0 + t * 16 + n] = acc[s2][t][r] + bv;
            }
        }
    }
}

// ---------------------------------------------------------------- launch
extern "C" void kernel_launch(void* const* d_in, const int* in_sizes, int n_in,
                              void* d_out, int out_size, void* d_ws, size_t ws_size,
                              hipStream_t stream) {
    const float* x = (const float*)d_in[0];
    const unsigned char* mask = (const unsigned char*)d_in[1];  // jnp bool = 1 byte
    const float* w_qkv = (const float*)d_in[2];
    const float* b_qkv = (const float*)d_in[3];
    const float* w_out = (const float*)d_in[4];
    const float* b_out = (const float*)d_in[5];
    float* out = (float*)d_out;

    char* ws = (char*)d_ws;
    u16b* xbf    = (u16b*)(ws);               // 4096x1024 bf16  (8 MB)
    u16b* wqkvbf = (u16b*)(ws + 8388608);     // 3072x1024 bf16  (6 MB)
    u16b* woutbf = (u16b*)(ws + 14680064);    // 1024x1024 bf16  (2 MB)
    u16b* Qb     = (u16b*)(ws + 16777216);    // [B,H,S,64] bf16 (8 MB)
    u16b* Kb     = (u16b*)(ws + 25165824);    // (8 MB)
    u16b* Vb     = (u16b*)(ws + 33554432);    // (8 MB)
    u16b* attnb  = (u16b*)(ws + 41943040);    // [B,S,1024] bf16 (8 MB)

    cvt_f32_bf16<<<4096, 256, 0, stream>>>(x, xbf, 4096 * 1024 / 4);
    cvt_f32_bf16<<<3072, 256, 0, stream>>>(w_qkv, wqkvbf, 3072 * 1024 / 4);
    cvt_f32_bf16<<<1024, 256, 0, stream>>>(w_out, woutbf, 1024 * 1024 / 4);

    gemm_qkv_rope<<<dim3(48, 16), 256, 0, stream>>>(xbf, wqkvbf, b_qkv, Qb, Kb, Vb);
    attention<<<dim3(16, 16, 2), 256, 0, stream>>>(Qb, Kb, Vb, mask, attnb);
    gemm_out<<<dim3(16, 16), 256, 0, stream>>>(attnb, woutbf, b_out, out);
}
